// GlobalHTADecoder_71399536329128
// MI455X (gfx1250) — compile-verified, matrix operand order checked by reading, not run on hardware
//
#include <hip/hip_runtime.h>
#include <hip/hip_bf16.h>

// ---------------------------------------------------------------------------
// HTA decoder for MI455X (gfx1250): f16 WMMA (v_wmma_f32_16x16x32_f16) GEMMs
// for all projections / attention matmuls, f32 softmax + layernorm.
// GEMM: block = 8 waves covering a 256(M)x64(N) macro tile; the shared 64-col
// B panel is double-buffered in LDS via GLOBAL_LOAD_ASYNC_TO_LDS_B128
// (ASYNCcnt), A streams from global; each wave runs a 32x64 register tile
// (8 WMMAs per K=32 step). Workspace: ~64 MB in d_ws.
// ---------------------------------------------------------------------------

#define D_   256
#define NH_  4
#define NL_  8
#define DFF_ 1024
#define VOC_ 100
#define LT_  256
#define BB_  2
#define FH_  32
#define FW_  256
#define S_   (FH_*FW_)   // 8192 cross-attn memory length
#define M_   (LT_*BB_)   // 512 token rows (r = l*B + b)
#define HD_  64
#define BD_  (BB_*D_)    // 512

#define LROW_ 72         // padded LDS row pitch in halves (144 B, 16B aligned)

typedef __attribute__((ext_vector_type(16))) _Float16 v16h;
typedef __attribute__((ext_vector_type(8)))  _Float16 v8h;
typedef __attribute__((ext_vector_type(8)))  float    v8f;

// build one 16-lane-striped v16h fragment from two 16B chunks 32B apart
__device__ __forceinline__ v16h frag_ld(const _Float16* p)
{
    v8h lo = *(const v8h*)(p);
    v8h hi = *(const v8h*)(p + 16);
    v16h r;
#pragma unroll
    for (int e = 0; e < 8; ++e) { r[e] = lo[e]; r[8 + e] = hi[e]; }
    return r;
}

// async copy 16 bytes global -> LDS (tracked by ASYNCcnt)
__device__ __forceinline__ void async_ld16(unsigned lds_off, const void* gaddr)
{
    asm volatile("global_load_async_to_lds_b128 %0, %1, off"
                 :: "v"(lds_off), "v"(gaddr) : "memory");
}
__device__ __forceinline__ void wait_async0()
{
    asm volatile("s_wait_asynccnt 0x0" ::: "memory");
}

// ------------------------------- GEMM (NT) ---------------------------------
// C[M,N] = A[M,K] * W[N,K]^T.  Requires M%256==0, N%64==0, K%64==0 (or K==64).
// Block: 256 threads = 8 waves; macro tile 256x64; wave w -> rows w*32..+32.
// B panel (64 rows x 64 halves, padded) double-buffered in LDS via async copy.
__global__ void gemm_nt_wmma(int M, int N, int K,
                             const _Float16* __restrict__ A, long lda,
                             const _Float16* __restrict__ W, long ldw,
                             const float* __restrict__ bias,
                             float* __restrict__ Cf,
                             _Float16* __restrict__ C16, long ldc,
                             float scale, int relu)
{
    __shared__ __align__(16) _Float16 lbuf[2][64 * LROW_];

    int tn = N >> 6;
    int bi = blockIdx.x / tn;                    // macro-tile row (256 rows)
    int bj = blockIdx.x % tn;                    // macro-tile col (64 cols)
    int wv   = threadIdx.x >> 5;                 // wave in block: row sub-tile
    int lane = threadIdx.x & 31;
    int half  = lane >> 4;                       // K-half selector
    int idx16 = lane & 15;                       // A row (m) / B col (n) in tile

    const _Float16* arow0 = A + (long)(bi * 256 + wv * 32 + idx16) * lda;
    const _Float16* arow1 = arow0 + 16 * lda;
    const _Float16* wpan  = W + (long)(bj * 64) * ldw;

    // stage one 64-row x 64-half B panel chunk into LDS buffer `bsel`
    auto stage = [&](int c, int bsel) {
#pragma unroll
        for (int t = 0; t < 2; ++t) {
            int s   = (int)threadIdx.x + t * 256;      // 512 x 16B segments
            int row = s >> 3, sir = s & 7;
            const _Float16* src = wpan + (long)row * ldw + (long)c * 64 + sir * 8;
            unsigned dst = (unsigned)(size_t)&lbuf[bsel][row * LROW_ + sir * 8];
            async_ld16(dst, src);
        }
    };

    v8f acc[2][4];
#pragma unroll
    for (int i = 0; i < 2; ++i)
#pragma unroll
        for (int j = 0; j < 4; ++j) acc[i][j] = (v8f){};

    int nc = K >> 6;                             // 64-half K chunks
    stage(0, 0);
    for (int c = 0; c < nc; ++c) {
        wait_async0();                           // own async stores landed
        __syncthreads();                         // whole panel visible
        if (c + 1 < nc) stage(c + 1, (c + 1) & 1);

        const _Float16* lb = &lbuf[c & 1][0];
#pragma unroll
        for (int kk = 0; kk < 2; ++kk) {         // two K=32 WMMA steps
            long kb  = (long)c * 64 + kk * 32 + half * 8;
            int  lkb =               kk * 32 + half * 8;
            __builtin_prefetch(arow0 + kb + 128, 0, 1);   // global_prefetch_b8

            v16h a0 = frag_ld(arow0 + kb);
            v16h a1 = frag_ld(arow1 + kb);
            v16h b0 = frag_ld(lb + (idx16)      * LROW_ + lkb);
            v16h b1 = frag_ld(lb + (16 + idx16) * LROW_ + lkb);
            v16h b2 = frag_ld(lb + (32 + idx16) * LROW_ + lkb);
            v16h b3 = frag_ld(lb + (48 + idx16) * LROW_ + lkb);

            acc[0][0] = __builtin_amdgcn_wmma_f32_16x16x32_f16(false, a0, false, b0, (short)0, acc[0][0], false, false);
            acc[0][1] = __builtin_amdgcn_wmma_f32_16x16x32_f16(false, a0, false, b1, (short)0, acc[0][1], false, false);
            acc[0][2] = __builtin_amdgcn_wmma_f32_16x16x32_f16(false, a0, false, b2, (short)0, acc[0][2], false, false);
            acc[0][3] = __builtin_amdgcn_wmma_f32_16x16x32_f16(false, a0, false, b3, (short)0, acc[0][3], false, false);
            acc[1][0] = __builtin_amdgcn_wmma_f32_16x16x32_f16(false, a1, false, b0, (short)0, acc[1][0], false, false);
            acc[1][1] = __builtin_amdgcn_wmma_f32_16x16x32_f16(false, a1, false, b1, (short)0, acc[1][1], false, false);
            acc[1][2] = __builtin_amdgcn_wmma_f32_16x16x32_f16(false, a1, false, b2, (short)0, acc[1][2], false, false);
            acc[1][3] = __builtin_amdgcn_wmma_f32_16x16x32_f16(false, a1, false, b3, (short)0, acc[1][3], false, false);
        }
    }

    // C/D layout: VGPR r -> M = r + 8*(lane/16); N = lane&15
#pragma unroll
    for (int i = 0; i < 2; ++i) {
#pragma unroll
        for (int j = 0; j < 4; ++j) {
            int nbase = bj * 64 + j * 16 + idx16;
            float badd = bias ? bias[nbase] : 0.0f;
#pragma unroll
            for (int r = 0; r < 8; ++r) {
                int m = bi * 256 + wv * 32 + i * 16 + r + 8 * half;
                float v = (acc[i][j][r] + badd) * scale;
                if (relu) v = fmaxf(v, 0.0f);
                long off = (long)m * ldc + nbase;
                if (Cf)  Cf[off]  = v;
                if (C16) C16[off] = (_Float16)v;
            }
        }
    }
}

// ------------------------------ helpers ------------------------------------
__global__ void cvt_f32_f16(const float* __restrict__ in,
                            _Float16* __restrict__ out, long n)
{
    long i = (long)blockIdx.x * blockDim.x + threadIdx.x;
    if (i < n) out[i] = (_Float16)in[i];
}

// mem16[(b*S + hw)*D + c] = features[b,c,h,w] + PE2D(c,h,w)  (f16)
__global__ void build_mem_kernel(const float* __restrict__ feat,
                                 _Float16* __restrict__ mem16)
{
    long idx = (long)blockIdx.x * blockDim.x + threadIdx.x;
    if (idx >= (long)BB_ * S_ * D_) return;
    int  c  = (int)(idx & (D_ - 1));
    long t  = idx >> 8;
    int  hw = (int)(t % S_);
    int  b  = (int)(t / S_);
    int  h  = hw / FW_, w = hw % FW_;
    float ang;
    if (c < D_ / 2) { int k = c >> 1;          ang = (float)h * __powf(10000.f, -(2.f * k) / (float)D_); }
    else            { int k = (c - D_/2) >> 1; ang = (float)w * __powf(10000.f, -(2.f * k) / (float)D_); }
    float pe = (c & 1) ? __cosf(ang) : __sinf(ang);
    float v  = feat[((long)b * D_ + c) * (long)S_ + hw] + pe;
    mem16[idx] = (_Float16)v;
}

// tgt[r, d] = emb[tokens[r], d] + PE1D(d, l)   (r = l*B + b) ; f32 + f16
__global__ void embed_kernel(const int* __restrict__ tokens,
                             const float* __restrict__ emb,
                             float* __restrict__ tgt32,
                             _Float16* __restrict__ tgt16)
{
    int r = blockIdx.x, d = threadIdx.x;
    int l = r / BB_;
    int tok = tokens[r];
    int k = d >> 1;
    float ang = (float)l * __powf(10000.f, -(2.f * k) / (float)D_);
    float pe  = (d & 1) ? __cosf(ang) : __sinf(ang);
    float v = emb[(long)tok * D_ + d] + pe;
    long off = (long)r * D_ + d;
    tgt32[off] = v;
    tgt16[off] = (_Float16)v;
}

// softmax over one score row (optionally causal); writes f16 probabilities
__global__ void softmax_kernel(const float* __restrict__ scores,
                               _Float16* __restrict__ p, int S, int causal)
{
    __shared__ float red[256];
    int l = blockIdx.x;
    const float* row = scores + (long)l * S;
    _Float16* prow   = p      + (long)l * S;
    int limit = causal ? (l + 1) : S;

    float m = -1e30f;
    for (int s = threadIdx.x; s < limit; s += blockDim.x) m = fmaxf(m, row[s]);
    red[threadIdx.x] = m; __syncthreads();
    for (int o = 128; o > 0; o >>= 1) {
        if ((int)threadIdx.x < o) red[threadIdx.x] = fmaxf(red[threadIdx.x], red[threadIdx.x + o]);
        __syncthreads();
    }
    m = red[0]; __syncthreads();

    float sum = 0.f;
    for (int s = threadIdx.x; s < limit; s += blockDim.x) sum += __expf(row[s] - m);
    red[threadIdx.x] = sum; __syncthreads();
    for (int o = 128; o > 0; o >>= 1) {
        if ((int)threadIdx.x < o) red[threadIdx.x] += red[threadIdx.x + o];
        __syncthreads();
    }
    float inv = 1.0f / red[0];
    for (int s = threadIdx.x; s < S; s += blockDim.x) {
        float v = (s < limit) ? __expf(row[s] - m) * inv : 0.f;
        prow[s] = (_Float16)v;
    }
}

// vt[d*S + s] = v[s*rowStride + d]   (d < 64) — turns P*V into NT form
__global__ void transpose_v_kernel(const _Float16* __restrict__ v, long rowStride,
                                   _Float16* __restrict__ vt, int S)
{
    long idx = (long)blockIdx.x * blockDim.x + threadIdx.x;
    if (idx >= (long)HD_ * S) return;
    int s = (int)(idx % S);
    int d = (int)(idx / S);
    vt[idx] = v[(long)s * rowStride + d];
}

// out = LayerNorm(x + t2) * g + b ; writes f32 + f16 (block = one row of 256)
__global__ void add_ln_kernel(const float* __restrict__ x,
                              const float* __restrict__ t2,
                              const float* __restrict__ g,
                              const float* __restrict__ b,
                              float* __restrict__ out32,
                              _Float16* __restrict__ out16)
{
    __shared__ float red[256];
    int r = blockIdx.x, d = threadIdx.x;
    long off = (long)r * D_ + d;
    float v = x[off] + t2[off];
    red[d] = v; __syncthreads();
    for (int o = 128; o > 0; o >>= 1) { if (d < o) red[d] += red[d + o]; __syncthreads(); }
    float mean = red[0] * (1.f / D_);
    __syncthreads();
    float c = v - mean;
    red[d] = c * c; __syncthreads();
    for (int o = 128; o > 0; o >>= 1) { if (d < o) red[d] += red[d + o]; __syncthreads(); }
    float var = red[0] * (1.f / D_);
    float y = c * rsqrtf(var + 1e-5f) * g[d] + b[d];
    out32[off] = y;
    out16[off] = (_Float16)y;
}

// logits[r, v] = dot(tgt32[r,:], out_w[v,:]) + out_b[v]  (V=100, plain f32)
__global__ void logits_kernel(const float* __restrict__ x,
                              const float* __restrict__ w,
                              const float* __restrict__ bias,
                              float* __restrict__ out)
{
    int r = blockIdx.x, v = threadIdx.x;
    if (v >= VOC_) return;
    const float* xr = x + (long)r * D_;
    const float* wr = w + (long)v * D_;
    float acc = bias[v];
    for (int i = 0; i < D_; ++i) acc += xr[i] * wr[i];
    out[(long)r * VOC_ + v] = acc;
}

// ------------------------------- launcher ----------------------------------
extern "C" void kernel_launch(void* const* d_in, const int* in_sizes, int n_in,
                              void* d_out, int out_size, void* d_ws, size_t ws_size,
                              hipStream_t stream)
{
    (void)in_sizes; (void)n_in; (void)out_size; (void)ws_size;

    const int*   tokens = (const int*)  d_in[0];
    const float* feats  = (const float*)d_in[1];
    const float* emb    = (const float*)d_in[2];
    const float* sa_w   = (const float*)d_in[3];
    const float* sa_b   = (const float*)d_in[4];
    const float* ca_w   = (const float*)d_in[5];
    const float* ca_b   = (const float*)d_in[6];
    const float* ln_g   = (const float*)d_in[7];
    const float* ln_b   = (const float*)d_in[8];
    const float* ff_w1  = (const float*)d_in[9];
    const float* ff_b1  = (const float*)d_in[10];
    const float* ff_w2  = (const float*)d_in[11];
    const float* ff_b2  = (const float*)d_in[12];
    const float* out_w  = (const float*)d_in[13];
    const float* out_b  = (const float*)d_in[14];
    float* logits = (float*)d_out;

    // ---- workspace arena (~60 MB) ----
    char* base = (char*)d_ws;
    size_t off = 0;
    auto alloc = [&](size_t bytes) -> void* {
        void* p = base + off;
        off = (off + bytes + 255) & ~(size_t)255;
        return p;
    };
    _Float16* mem16  = (_Float16*)alloc((size_t)BB_ * S_ * D_ * 2);
    float*    tgt32  = (float*)   alloc((size_t)M_ * D_ * 4);
    _Float16* tgt16  = (_Float16*)alloc((size_t)M_ * D_ * 2);
    _Float16* q16    = (_Float16*)alloc((size_t)M_ * D_ * 2);
    _Float16* k16    = (_Float16*)alloc((size_t)M_ * D_ * 2);
    _Float16* v16    = (_Float16*)alloc((size_t)M_ * D_ * 2);
    _Float16* kc16   = (_Float16*)alloc((size_t)BB_ * S_ * D_ * 2);
    _Float16* vc16   = (_Float16*)alloc((size_t)BB_ * S_ * D_ * 2);
    _Float16* o16    = (_Float16*)alloc((size_t)M_ * D_ * 2);
    float*    t2     = (float*)   alloc((size_t)M_ * D_ * 4);
    _Float16* h16    = (_Float16*)alloc((size_t)M_ * DFF_ * 2);
    float*    sc_u   = (float*)   alloc((size_t)LT_ * S_ * 4);   // per-unit scores
    _Float16* p16_u  = (_Float16*)alloc((size_t)LT_ * S_ * 2);   // per-unit probs
    _Float16* vt_u   = (_Float16*)alloc((size_t)HD_ * S_ * 2);   // per-unit V^T
    _Float16* sa16   = (_Float16*)alloc((size_t)NL_ * 4 * D_ * D_ * 2);
    _Float16* ca16   = (_Float16*)alloc((size_t)NL_ * 4 * D_ * D_ * 2);
    _Float16* w1_16  = (_Float16*)alloc((size_t)NL_ * DFF_ * D_ * 2);
    _Float16* w2_16  = (_Float16*)alloc((size_t)NL_ * D_ * DFF_ * 2);

    auto gemm = [&](int M, int N, int K,
                    const _Float16* A, long lda, const _Float16* W, long ldw,
                    const float* bias, float* Cf, _Float16* C16, long ldc,
                    float scale, int relu) {
        int blocks = (M / 256) * (N / 64);          // 256x64 macro tile / block
        gemm_nt_wmma<<<dim3(blocks), dim3(256), 0, stream>>>(
            M, N, K, A, lda, W, ldw, bias, Cf, C16, ldc, scale, relu);
    };
    auto cvt = [&](const float* in, _Float16* out, long n) {
        cvt_f32_f16<<<dim3((int)((n + 255) / 256)), dim3(256), 0, stream>>>(in, out, n);
    };

    // ---- one-time prep ----
    cvt(sa_w,  sa16,  (long)NL_ * 4 * D_ * D_);
    cvt(ca_w,  ca16,  (long)NL_ * 4 * D_ * D_);
    cvt(ff_w1, w1_16, (long)NL_ * DFF_ * D_);
    cvt(ff_w2, w2_16, (long)NL_ * D_ * DFF_);
    build_mem_kernel<<<dim3((int)(((long)BB_ * S_ * D_ + 255) / 256)), dim3(256), 0, stream>>>(feats, mem16);
    embed_kernel<<<dim3(M_), dim3(D_), 0, stream>>>(tokens, emb, tgt32, tgt16);

    const float qs = 0.125f;   // hd^-0.5, hd = 64

    for (int i = 0; i < NL_; ++i) {
        // ======================= self-attention =======================
        const _Float16* Wq = sa16 + (size_t)(i * 4 + 0) * D_ * D_;
        const _Float16* Wk = sa16 + (size_t)(i * 4 + 1) * D_ * D_;
        const _Float16* Wv = sa16 + (size_t)(i * 4 + 2) * D_ * D_;
        const _Float16* Wo = sa16 + (size_t)(i * 4 + 3) * D_ * D_;
        const float* bq = sa_b + (size_t)(i * 4 + 0) * D_;
        const float* bk = sa_b + (size_t)(i * 4 + 1) * D_;
        const float* bv = sa_b + (size_t)(i * 4 + 2) * D_;
        const float* bo = sa_b + (size_t)(i * 4 + 3) * D_;

        gemm(M_, D_, D_, tgt16, D_, Wq, D_, bq, nullptr, q16, D_, qs, 0);
        gemm(M_, D_, D_, tgt16, D_, Wk, D_, bk, nullptr, k16, D_, 1.f, 0);
        gemm(M_, D_, D_, tgt16, D_, Wv, D_, bv, nullptr, v16, D_, 1.f, 0);

        for (int b = 0; b < BB_; ++b)
            for (int h = 0; h < NH_; ++h) {
                const _Float16* Qh = q16 + b * D_ + h * HD_;   // row stride B*D
                const _Float16* Kh = k16 + b * D_ + h * HD_;
                const _Float16* Vh = v16 + b * D_ + h * HD_;
                gemm(LT_, LT_, HD_, Qh, BD_, Kh, BD_, nullptr, sc_u, nullptr, LT_, 1.f, 0);
                softmax_kernel<<<dim3(LT_), dim3(256), 0, stream>>>(sc_u, p16_u, LT_, 1);
                transpose_v_kernel<<<dim3((HD_ * LT_ + 255) / 256), dim3(256), 0, stream>>>(Vh, BD_, vt_u, LT_);
                gemm(LT_, HD_, LT_, p16_u, LT_, vt_u, LT_, nullptr, nullptr,
                     o16 + b * D_ + h * HD_, BD_, 1.f, 0);
            }
        gemm(M_, D_, D_, o16, D_, Wo, D_, bo, t2, nullptr, D_, 1.f, 0);
        add_ln_kernel<<<dim3(M_), dim3(D_), 0, stream>>>(
            tgt32, t2, ln_g + (size_t)(i * 3 + 0) * D_, ln_b + (size_t)(i * 3 + 0) * D_, tgt32, tgt16);

        // ======================= cross-attention ======================
        const _Float16* cWq = ca16 + (size_t)(i * 4 + 0) * D_ * D_;
        const _Float16* cWk = ca16 + (size_t)(i * 4 + 1) * D_ * D_;
        const _Float16* cWv = ca16 + (size_t)(i * 4 + 2) * D_ * D_;
        const _Float16* cWo = ca16 + (size_t)(i * 4 + 3) * D_ * D_;
        const float* cbq = ca_b + (size_t)(i * 4 + 0) * D_;
        const float* cbk = ca_b + (size_t)(i * 4 + 1) * D_;
        const float* cbv = ca_b + (size_t)(i * 4 + 2) * D_;
        const float* cbo = ca_b + (size_t)(i * 4 + 3) * D_;

        gemm(M_, D_, D_, tgt16, D_, cWq, D_, cbq, nullptr, q16, D_, qs, 0);
        gemm(BB_ * S_, D_, D_, mem16, D_, cWk, D_, cbk, nullptr, kc16, D_, 1.f, 0);
        gemm(BB_ * S_, D_, D_, mem16, D_, cWv, D_, cbv, nullptr, vc16, D_, 1.f, 0);

        for (int b = 0; b < BB_; ++b)
            for (int h = 0; h < NH_; ++h) {
                const _Float16* Qh = q16  + b * D_ + h * HD_;              // row stride B*D
                const _Float16* Kh = kc16 + (size_t)b * S_ * D_ + h * HD_; // row stride D
                const _Float16* Vh = vc16 + (size_t)b * S_ * D_ + h * HD_;
                gemm(LT_, S_, HD_, Qh, BD_, Kh, D_, nullptr, sc_u, nullptr, S_, 1.f, 0);
                softmax_kernel<<<dim3(LT_), dim3(256), 0, stream>>>(sc_u, p16_u, S_, 0);
                transpose_v_kernel<<<dim3((HD_ * S_ + 255) / 256), dim3(256), 0, stream>>>(Vh, D_, vt_u, S_);
                gemm(LT_, HD_, S_, p16_u, S_, vt_u, S_, nullptr, nullptr,
                     o16 + b * D_ + h * HD_, BD_, 1.f, 0);
            }
        gemm(M_, D_, D_, o16, D_, cWo, D_, cbo, t2, nullptr, D_, 1.f, 0);
        add_ln_kernel<<<dim3(M_), dim3(D_), 0, stream>>>(
            tgt32, t2, ln_g + (size_t)(i * 3 + 1) * D_, ln_b + (size_t)(i * 3 + 1) * D_, tgt32, tgt16);

        // ============================ FFN =============================
        gemm(M_, DFF_, D_, tgt16, D_, w1_16 + (size_t)i * DFF_ * D_, D_,
             ff_b1 + (size_t)i * DFF_, nullptr, h16, DFF_, 1.f, 1);
        gemm(M_, D_, DFF_, h16, DFF_, w2_16 + (size_t)i * D_ * DFF_, DFF_,
             ff_b2 + (size_t)i * D_, t2, nullptr, D_, 1.f, 0);
        add_ln_kernel<<<dim3(M_), dim3(D_), 0, stream>>>(
            tgt32, t2, ln_g + (size_t)(i * 3 + 2) * D_, ln_b + (size_t)(i * 3 + 2) * D_, tgt32, tgt16);
    }

    // ---- final vocabulary projection (V=100, not tile friendly -> f32) ----
    logits_kernel<<<dim3(M_), dim3(128), 0, stream>>>(tgt32, out_w, out_b, logits);
}
